// Based_74783970558672
// MI455X (gfx1250) — matrix operand order repaired and checked by reading, not compiled, not run on hardware
//
#include <hip/hip_runtime.h>

typedef __bf16 bf16;
typedef __attribute__((ext_vector_type(16))) __bf16 v16bf;
typedef __attribute__((ext_vector_type(4)))  __bf16 v4bf;
typedef __attribute__((ext_vector_type(8)))  float  v8f;
typedef __attribute__((ext_vector_type(4)))  float  v4f;
typedef __attribute__((ext_vector_type(4)))  int    v4i;

#define BB 2
#define LL 2048
#define DD 1536
#define HH 12
#define FD 16
#define HD 128

// ---- CDNA5 async-to-LDS path (guarded; falls back to sync copies) ---------
#if defined(__AMDGCN__) && __has_builtin(__builtin_amdgcn_global_load_async_to_lds_b128) && __has_builtin(__builtin_amdgcn_s_wait_asynccnt)
#define USE_ASYNC 1
#else
#define USE_ASYNC 0
#endif

#if USE_ASYNC
typedef __attribute__((address_space(1))) v4i gas_v4i;   // global int4
typedef __attribute__((address_space(3))) v4i las_v4i;   // LDS int4
static __device__ __forceinline__ void async_cp16(const void* g, void* l) {
  // emits global_load_async_to_lds_b128 (ASYNCcnt-tracked DMA into LDS)
  __builtin_amdgcn_global_load_async_to_lds_b128(
      (gas_v4i*)g, (las_v4i*)l, 0, 0);
}
static __device__ __forceinline__ void async_wait0() {
  __builtin_amdgcn_s_wait_asynccnt(0);
}
#else
static __device__ __forceinline__ void async_cp16(const void* g, void* l) {
  *(v4f*)l = *(const v4f*)g;
}
static __device__ __forceinline__ void async_wait0() {}
#endif

static __device__ __forceinline__ v8f wmma_bf16(v16bf a, v16bf b, v8f c) {
  // D = A(16x32 bf16) x B(32x16 bf16) + C(16x16 f32)
  return __builtin_amdgcn_wmma_f32_16x16x32_bf16(
      /*neg_a=*/false, a, /*neg_b=*/false, b,
      /*c_mod=*/(short)0, c, /*reuse_a=*/false, /*reuse_b=*/false);
}

// ---------------------------------------------------------------------------
// GEMM: C[M x N] = A[M x K] * (B[K x N] * bscale)
// A element type AT in {float, bf16}; B is fp32, converted to bf16 in LDS.
// Block tile 128x64, 8 waves (256 threads), K-step 32.
// OUT_MODE: 0 = f32 row-major, 1 = bf16 row-major,
//           2 = bf16 "V-transposed": Vt[b][h][vdim][l]  (keys contiguous)
// ---------------------------------------------------------------------------
template<typename AT, int OUT_MODE>
__global__ __launch_bounds__(256)
void gemm_wmma(const AT* __restrict__ A, const float* __restrict__ Bm,
               void* __restrict__ Cout, int M, int N, int K, float bscale)
{
  (void)M;
  __shared__ bf16 sA[128][32];     // row-major tile of A (bf16)
  __shared__ bf16 sBt[64][32];     // B tile stored transposed: [n][k]

  const int tid  = threadIdx.x;
  const int wave = tid >> 5;
  const int lane = tid & 31;
  const int ml   = lane & 15;
  const int hs   = (lane >> 4) & 1;
  const int m0   = blockIdx.y * 128;
  const int n0   = blockIdx.x * 64;

  v8f acc[4];
  #pragma unroll
  for (int nt = 0; nt < 4; ++nt)
    #pragma unroll
    for (int j = 0; j < 8; ++j) acc[nt][j] = 0.0f;

  for (int k0 = 0; k0 < K; k0 += 32) {
    __syncthreads();
    // ---- stage A tile ----
    if constexpr (sizeof(AT) == 4) {
      #pragma unroll
      for (int i = 0; i < 4; ++i) {
        int idx = tid + i * 256;              // 1024 groups of 4 floats
        int r = idx >> 3, cg = (idx & 7) * 4;
        v4f v = *(const v4f*)&A[(size_t)(m0 + r) * K + k0 + cg];
        v4bf o;
        #pragma unroll
        for (int j = 0; j < 4; ++j) o[j] = (bf16)v[j];
        *(v4bf*)&sA[r][cg] = o;
      }
    } else {
      #pragma unroll
      for (int i = 0; i < 2; ++i) {
        int idx = tid + i * 256;              // 512 groups of 8 bf16
        int r = idx >> 2, cg = (idx & 3) * 8;
        async_cp16(&A[(size_t)(m0 + r) * K + k0 + cg], &sA[r][cg]);
      }
    }
    // ---- stage B tile transposed (fp32 -> bf16, scaled) ----
    {
      int kr = tid >> 3;                      // 0..31
      int ng = (tid & 7) * 8;
      const float* bp = &Bm[(size_t)(k0 + kr) * N + n0 + ng];
      v4f b0 = *(const v4f*)bp;
      v4f b1 = *(const v4f*)(bp + 4);
      #pragma unroll
      for (int j = 0; j < 4; ++j) sBt[ng + j][kr]     = (bf16)(b0[j] * bscale);
      #pragma unroll
      for (int j = 0; j < 4; ++j) sBt[ng + 4 + j][kr] = (bf16)(b1[j] * bscale);
    }
    if constexpr (sizeof(AT) == 2) async_wait0();
    __syncthreads();

    // A fragment: M = lane%16; elem e -> K = hs*16 + e
    union { v16bf v; v4f f[2]; } ua;
    const bf16* ap = &sA[(wave << 4) | ml][hs * 16];
    ua.f[0] = *(const v4f*)ap;
    ua.f[1] = *(const v4f*)(ap + 8);

    #pragma unroll
    for (int nt = 0; nt < 4; ++nt) {
      // B fragment: N = lane%16; elem e -> K = hs*16 + e (contiguous in sBt row)
      union { v16bf v; v4f f[2]; } ub;
      const bf16* bp = &sBt[(nt << 4) | ml][hs * 16];
      ub.f[0] = *(const v4f*)bp;
      ub.f[1] = *(const v4f*)(bp + 8);
      acc[nt] = wmma_bf16(ua.v, ub.v, acc[nt]);
    }
  }

  // C fragment: elem j -> row = hs*8 + j, col = lane%16
  #pragma unroll
  for (int nt = 0; nt < 4; ++nt) {
    #pragma unroll
    for (int j = 0; j < 8; ++j) {
      size_t row = (size_t)m0 + (wave << 4) + hs * 8 + j;
      size_t col = (size_t)n0 + (nt << 4) + ml;
      float v = acc[nt][j];
      if constexpr (OUT_MODE == 0) {
        ((float*)Cout)[row * N + col] = v;
      } else if constexpr (OUT_MODE == 1) {
        ((bf16*)Cout)[row * N + col] = (bf16)v;
      } else {
        // transposed V store: row = b*LL + l, col = h*HD + vd
        size_t bb2 = row >> 11;           // / LL
        size_t l2  = row & (LL - 1);
        size_t h2  = col >> 7;            // / HD
        size_t vd2 = col & (HD - 1);
        ((bf16*)Cout)[((bb2 * HH + h2) * HD + vd2) * LL + l2] = (bf16)v;
      }
    }
  }
}

// ---------------------------------------------------------------------------
// Attention helpers
// ---------------------------------------------------------------------------
// Stage one 64-key chunk: K rows (16 real features) + Vt rows (vdim-major,
// 64 contiguous keys per row). All copies are b128 (async when enabled).
static __device__ __forceinline__ void stage_chunk(
    const bf16* __restrict__ Kg,      // K base at this chunk's first row
    const bf16* __restrict__ Vtg,     // Vt base for (b,h) at this chunk's col
    bf16 (*sKb)[32], bf16 (*sVtb)[64], int tid)
{
  const int kr = tid >> 1, half = tid & 1;
  async_cp16(Kg + (size_t)kr * (HH * FD) + half * 8, &sKb[kr][half * 8]);
  const int vd = tid;                           // 0..127
  const bf16* vp = Vtg + (size_t)vd * LL;
  bf16* vdst = sVtb[vd];
  #pragma unroll
  for (int g = 0; g < 8; ++g) async_cp16(vp + g * 8, vdst + g * 8);
}

// S = q k^T over 4 n-tiles, Taylor map (+ causal mask on the diagonal chunk),
// accumulate normalizer, stash bf16 scores into this wave's sAm rows.
template<bool MASK>
static __device__ __forceinline__ void score_tiles(
    v16bf qv, const bf16 (*sKb)[32], bf16 (*sAm)[64], float* zp,
    int wave, int ml, int hs, int kc, int q0)
{
  v8f czero;
  #pragma unroll
  for (int j = 0; j < 8; ++j) czero[j] = 0.0f;
  #pragma unroll
  for (int nt = 0; nt < 4; ++nt) {
    union { v16bf v; v4f f[2]; } kb;
    const bf16* bp = &sKb[(nt << 4) | ml][hs * 8];
    kb.f[0] = *(const v4f*)bp;          // elems 8..15 read the zero pad
    kb.f[1] = *(const v4f*)(bp + 16);
    v8f s = wmma_bf16(qv, kb.v, czero);
    #pragma unroll
    for (int j = 0; j < 8; ++j) {
      float sv = s[j];
      float a = 1.0f + sv + 0.5f * sv * sv;
      if (MASK) {
        int kg = kc * 64 + (nt << 4) + ml;
        int qr = q0 + hs * 8 + j;
        if (kg > qr) a = 0.0f;
      }
      zp[j] += a;
      sAm[(wave << 4) + hs * 8 + j][(nt << 4) + ml] = (bf16)a;
    }
  }
}

// ---------------------------------------------------------------------------
// Causal "based" attention: 4 waves, 64 query rows / block, 64-key chunks,
// double-buffered async K/Vt staging overlapped with WMMA compute.
// ---------------------------------------------------------------------------
__global__ __launch_bounds__(128)
void based_attn(const bf16* __restrict__ Q, const bf16* __restrict__ Kk,
                const bf16* __restrict__ Vt, bf16* __restrict__ O)
{
  __shared__ bf16 sK[2][64][32];    // key chunk, features padded to 32 (8 KB)
  __shared__ bf16 sVt[2][128][64];  // Vt chunk: [vdim][key] (32 KB)
  __shared__ bf16 sAm[64][64];      // bf16 Taylor scores, per-wave rows (8 KB)

  const int tid  = threadIdx.x;
  const int wave = tid >> 5;
  const int lane = tid & 31;
  const int ml   = lane & 15;
  const int hs   = (lane >> 4) & 1;
  const int qb = blockIdx.x;
  const int h  = blockIdx.y;
  const int b  = blockIdx.z;
  const int q0 = qb * 64 + wave * 16;

  const bf16* Kbase  = Kk + (size_t)b * LL * (HH * FD) + h * FD;
  const bf16* Vtbase = Vt + ((size_t)b * HH + h) * HD * LL;

  v4f zero4;
  #pragma unroll
  for (int j = 0; j < 4; ++j) zero4[j] = 0.0f;

  // one-time zero pad of K feature cols 16..31 (never rewritten)
  {
    int kr = tid >> 1, half = tid & 1;
    *(v4f*)&sK[0][kr][16 + half * 8] = zero4;
    *(v4f*)&sK[1][kr][16 + half * 8] = zero4;
  }

  // q fragment (loaded once; K elems >= 16 are zero)
  union { v16bf v; v4f f[2]; } qa;
  {
    const bf16* qp = &Q[((size_t)b * LL + q0 + ml) * (HH * FD) + h * FD + hs * 8];
    qa.f[0] = *(const v4f*)qp;
    qa.f[1] = zero4;
  }

  v8f oacc[8];
  #pragma unroll
  for (int nt = 0; nt < 8; ++nt)
    #pragma unroll
    for (int j = 0; j < 8; ++j) oacc[nt][j] = 0.0f;
  float zp[8];
  #pragma unroll
  for (int j = 0; j < 8; ++j) zp[j] = 0.0f;

  // prologue: stage chunk 0 into buffer 0
  stage_chunk(Kbase, Vtbase, sK[0], sVt[0], tid);

  for (int kc = 0; kc <= qb; ++kc) {
    const int cur = kc & 1;
    async_wait0();                 // my wave's staging of `cur` complete
    __syncthreads();               // everyone's staging of `cur` complete
    if (kc < qb)                   // overlap: DMA next chunk into other buffer
      stage_chunk(Kbase + (size_t)(kc + 1) * 64 * (HH * FD),
                  Vtbase + (size_t)(kc + 1) * 64,
                  sK[cur ^ 1], sVt[cur ^ 1], tid);

    if (kc < qb) score_tiles<false>(qa.v, sK[cur], sAm, zp, wave, ml, hs, kc, q0);
    else         score_tiles<true >(qa.v, sK[cur], sAm, zp, wave, ml, hs, kc, q0);

    // O += A @ V  (own 16 rows of sAm only -> no inter-wave barrier needed)
    #pragma unroll
    for (int kk = 0; kk < 2; ++kk) {
      union { v16bf v; v4f f[2]; } aa;
      const bf16* ap = &sAm[(wave << 4) | ml][kk * 32 + hs * 16];
      aa.f[0] = *(const v4f*)ap;
      aa.f[1] = *(const v4f*)(ap + 8);
      #pragma unroll
      for (int nt = 0; nt < 8; ++nt) {
        // B fragment: N = lane%16 fixed; K contiguous in sVt row -> 2x b128
        union { v16bf v; v4f f[2]; } vb;
        const bf16* vp2 = &sVt[cur][(nt << 4) | ml][kk * 32 + hs * 16];
        vb.f[0] = *(const v4f*)vp2;
        vb.f[1] = *(const v4f*)(vp2 + 8);
        oacc[nt] = wmma_bf16(aa.v, vb.v, oacc[nt]);
      }
    }
  }

  // finish normalizer: row z lives across the 16 lanes of each half-wave
  #pragma unroll
  for (int j = 0; j < 8; ++j) {
    float z = zp[j];
    z += __shfl_xor(z, 1, 32);
    z += __shfl_xor(z, 2, 32);
    z += __shfl_xor(z, 4, 32);
    z += __shfl_xor(z, 8, 32);
    float r = 1.0f / (z + 1e-12f);
    #pragma unroll
    for (int nt = 0; nt < 8; ++nt) oacc[nt][j] *= r;
  }

  // store O (bf16) in [B*L, H*HD] layout for the Wo GEMM
  #pragma unroll
  for (int nt = 0; nt < 8; ++nt) {
    #pragma unroll
    for (int j = 0; j < 8; ++j) {
      size_t l = (size_t)q0 + hs * 8 + j;
      size_t c = (size_t)h * HD + (nt << 4) + ml;
      O[((size_t)b * LL + l) * (HH * HD) + c] = (bf16)oacc[nt][j];
    }
  }
}

// ---------------------------------------------------------------------------
extern "C" void kernel_launch(void* const* d_in, const int* in_sizes, int n_in,
                              void* d_out, int out_size, void* d_ws, size_t ws_size,
                              hipStream_t stream)
{
  (void)in_sizes; (void)n_in; (void)out_size; (void)ws_size;
  const float* X  = (const float*)d_in[0];
  const float* Wq = (const float*)d_in[1];
  const float* Wk = (const float*)d_in[2];
  const float* Wv = (const float*)d_in[3];
  const float* Wo = (const float*)d_in[4];

  const size_t BL = (size_t)BB * LL;      // 4096
  bf16* Qb  = (bf16*)d_ws;                // [4096 x 192]
  bf16* Kb  = Qb + BL * HH * FD;          // [4096 x 192]
  bf16* Vtb = Kb + BL * HH * FD;          // [B][H][128][2048] transposed V
  bf16* Ob  = Vtb + BL * DD;              // [4096 x 1536]   (~28.3 MB total)

  // q/k/v projections (q-scale 16^-0.5 = 0.25 folded into Wq conversion)
  gemm_wmma<float, 1><<<dim3((HH*FD)/64, BL/128), 256, 0, stream>>>(
      X, Wq, Qb, (int)BL, HH*FD, DD, 0.25f);
  gemm_wmma<float, 1><<<dim3((HH*FD)/64, BL/128), 256, 0, stream>>>(
      X, Wk, Kb, (int)BL, HH*FD, DD, 1.0f);
  gemm_wmma<float, 2><<<dim3(DD/64, BL/128), 256, 0, stream>>>(
      X, Wv, Vtb, (int)BL, DD, DD, 1.0f);

  // causal Taylor attention
  based_attn<<<dim3(LL/64, HH, BB), 128, 0, stream>>>(Qb, Kb, Vtb, Ob);

  // output projection -> fp32
  gemm_wmma<bf16, 0><<<dim3(DD/64, BL/128), 256, 0, stream>>>(
      Ob, Wo, d_out, (int)BL, DD, DD, 1.0f);
}
